// MonotonicModel_v1_90065464197296
// MI455X (gfx1250) — matrix operand order repaired and checked by reading, not compiled, not run on hardware
//
#include <hip/hip_runtime.h>
#include <hip/hip_bf16.h>
#include <math.h>

typedef __attribute__((ext_vector_type(16))) _Float16 v16h;
typedef __attribute__((ext_vector_type(8)))  float    v8f;

#define HIDN 256
#define NB_STEPS 32
#define SSTEPS (NB_STEPS + 1)          /* 33 quadrature nodes            */
#define NROWS 1024
#define NCOLS 10
#define NX (NROWS * NCOLS)             /* 10240 evaluation points        */
#define MTOTAL (NX * SSTEPS)           /* 337920 MLP rows                */
#define WAVES 16
#define BLOCK_THREADS (WAVES * 32)     /* 512 threads = 16 waves         */
#define ROWS_PER_BLOCK (WAVES * 16)    /* 256 rows per block-tile        */
#define NUM_TILES (MTOTAL / ROWS_PER_BLOCK) /* 1320 exactly              */
#define MAIN_GRID 330
#define PREP_THREADS 1024

#define SMEM_W2_HALFS (HIDN * HIDN)    /* 65536 f16 = 128 KB             */
#define SMEM_W2_BYTES (SMEM_W2_HALFS * 2)
#define SMEM_BYTES (SMEM_W2_BYTES + 4 * HIDN * 4) /* 135168 B            */

__device__ __forceinline__ float elu1(float x) {
  return x > 0.0f ? x : (__expf(x) - 1.0f);
}

/* -------- kernel 1: log-softmax + CC quadrature + W2 f16 permute ------ */
__global__ void prep_kernel(const float* __restrict__ logits,
                            const float* __restrict__ W2,
                            float* __restrict__ xbuf,
                            float* __restrict__ ccw,
                            float* __restrict__ steps,
                            _Float16* __restrict__ w2h) {
  int row = blockIdx.x * blockDim.x + threadIdx.x;  /* 0..1023 */
  if (row < NROWS) {
    const float* l = logits + row * NCOLS;
    float m = -INFINITY;
    for (int j = 0; j < NCOLS; ++j) m = fmaxf(m, l[j]);
    float se = 0.0f;
    for (int j = 0; j < NCOLS; ++j) se += __expf(l[j] - m);
    float lse = m + __logf(se);
    for (int j = 0; j < NCOLS; ++j) xbuf[row * NCOLS + j] = l[j] - lse;
  }
  /* W2 (f32, row-major [k][n]) -> f16, pre-permuted to B-fragment order:
     tile (kk,nt) holds K=kk*32..+31, N=nt*16..+15; within a tile,
     lane = (K%32 >= 16)*16 + N%16, element e = K%16, one v16h per lane. */
  for (int idx = row; idx < HIDN * HIDN; idx += PREP_THREADS) {
    int k = idx >> 8;
    int n = idx & 255;
    int kk = k >> 5;
    int k5 = k & 31;
    int bhi = k5 >> 4;
    int e  = k5 & 15;
    int nt = n >> 4;
    int ln = (bhi << 4) | (n & 15);
    w2h[((kk * 16 + nt) * 512) + ln * 16 + e] = (_Float16)W2[idx];
  }
  /* Clenshaw-Curtis weights/nodes, exactly as _cc_quadrature (f64 math) */
  if (blockIdx.x == 0 && threadIdx.x < SSTEPS) {
    const double PI = 3.14159265358979323846;
    int j = threadIdx.x;
    double s = 0.0;
    for (int k = 0; k <= NB_STEPS; ++k) {
      double lam;
      if (j == 0) lam = 0.5;
      else {
        lam = cos((double)k * (double)j * PI / (double)NB_STEPS);
        if (j == NB_STEPS) lam *= 0.5;
      }
      lam *= 2.0 / (double)NB_STEPS;
      double wv;
      if (k & 1)       wv = 0.0;
      else if (k == 0) wv = 1.0;
      else             wv = 2.0 / (1.0 - (double)k * (double)k);
      s += lam * wv;
    }
    ccw[j]   = (float)s;
    steps[j] = (float)cos((double)j * PI / (double)NB_STEPS);
  }
}

/* ---------------- kernel 2: WMMA batched MLP ------------------------- */
__global__ __launch_bounds__(BLOCK_THREADS)
void umnn_main_kernel(const float* __restrict__ W1, const float* __restrict__ b1,
                      const _Float16* __restrict__ w2h, const float* __restrict__ b2,
                      const float* __restrict__ W3, const float* __restrict__ b3,
                      const float* __restrict__ xbuf,
                      const float* __restrict__ steps,
                      float* __restrict__ gbuf) {
  extern __shared__ __attribute__((aligned(128))) char smem[];
  float* sW1 = (float*)(smem + SMEM_W2_BYTES);
  float* sB1 = sW1 + HIDN;
  float* sB2 = sB1 + HIDN;
  float* sW3 = sB2 + HIDN;

  const int tid = threadIdx.x;

  /* Async DMA the pre-permuted f16 W2 (128 KB) into LDS: 16 B per lane per
     issue, GVS addressing (saddr base + 32-bit voffset), ASYNCcnt-tracked. */
  {
    const uint32_t ldsBase = (uint32_t)(uintptr_t)(void*)smem; /* addr[31:0] = LDS offset */
    for (int i = tid; i < SMEM_W2_BYTES / 16; i += BLOCK_THREADS) {
      uint32_t byteOff  = (uint32_t)i * 16u;
      uint32_t lds_addr = ldsBase + byteOff;
      asm volatile("global_load_async_to_lds_b128 %0, %1, %2"
                   :: "v"(lds_addr), "v"(byteOff), "s"(w2h)
                   : "memory");
    }
  }
  if (tid < HIDN) {
    sW1[tid] = W1[tid];
    sB1[tid] = b1[tid];
    sB2[tid] = b2[tid];
    sW3[tid] = W3[tid];
  }
#if __has_builtin(__builtin_amdgcn_s_wait_asynccnt)
  __builtin_amdgcn_s_wait_asynccnt(0);
#else
  asm volatile("s_wait_asynccnt 0x0" ::: "memory");
#endif
  __syncthreads();

  const int wid  = tid >> 5;
  const int lane = tid & 31;
  const int lrow = lane & 15;   /* data row within 16-row tile / C column */
  const int hi   = lane >> 4;
  const float b3v = b3[0];
  const v16h* bfr = (const v16h*)smem;

  for (int tile = blockIdx.x; tile < NUM_TILES; tile += gridDim.x) {
    const int rowBase = tile * ROWS_PER_BLOCK + wid * 16;
    const int r = rowBase + lrow;
    const int nIdx = r / SSTEPS;
    const int sIdx = r - nIdx * SSTEPS;
    const float x = xbuf[nIdx];
    const float t = x * (steps[sIdx] + 1.0f) * 0.5f;

    float partial[8];
    #pragma unroll
    for (int v = 0; v < 8; ++v) partial[v] = 0.0f;

    #pragma unroll
    for (int pass = 0; pass < 2; ++pass) {
      v8f acc[8];
      #pragma unroll
      for (int j = 0; j < 8; ++j) { v8f z = {}; acc[j] = z; }

      for (int kk = 0; kk < 8; ++kk) {
        /* A fragment: h1 = elu(t*W1+b1) built directly in the
           contiguous layout matching B: element e holds K = kk*32 + hi*16 + e. */
        v16h a;
        #pragma unroll
        for (int e = 0; e < 16; ++e) {
          int K = (kk << 5) + (hi << 4) + e;
          float h = fmaf(t, sW1[K], sB1[K]);
          a[e] = (_Float16)elu1(h);
        }
        #pragma unroll
        for (int j = 0; j < 8; ++j) {
          int nt = pass * 8 + j;
          v16h bb = bfr[(kk * 16 + nt) * 32 + lane];
          acc[j] = __builtin_amdgcn_wmma_f32_16x16x32_f16(
              false, a, false, bb, (short)0, acc[j], false, false);
        }
      }
      /* epilogue for this pass: bias + elu, then dot with W3 per-lane.
         C layout: acc[j][v] = h2[row = v + 8*hi][hid = (pass*8+j)*16 + lrow] */
      #pragma unroll
      for (int j = 0; j < 8; ++j) {
        int hidx = (pass * 8 + j) * 16 + lrow;
        float b2v = sB2[hidx];
        float w3v = sW3[hidx];
        #pragma unroll
        for (int v = 0; v < 8; ++v) {
          float h2 = elu1(acc[j][v] + b2v);
          partial[v] = fmaf(h2, w3v, partial[v]);
        }
      }
    }

    /* reduce the 16 columns held within each 16-lane half */
    #pragma unroll
    for (int m = 1; m <= 8; m <<= 1) {
      #pragma unroll
      for (int v = 0; v < 8; ++v)
        partial[v] += __shfl_xor(partial[v], m, 32);
    }

    /* lane (hi, lrow<8) publishes row hi*8 + lrow */
    float o = partial[0];
    #pragma unroll
    for (int v = 1; v < 8; ++v) if (lrow == v) o = partial[v];
    if (lrow < 8) {
      float g = elu1(o + b3v) + 1.0f;
      gbuf[rowBase + (hi << 3) + lrow] = g;
    }
  }
}

/* ---------------- kernel 3: quadrature reduction --------------------- */
__global__ void finish_kernel(const float* __restrict__ xbuf,
                              const float* __restrict__ gbuf,
                              const float* __restrict__ ccw,
                              const float* __restrict__ offset,
                              float* __restrict__ out) {
  int i = blockIdx.x * blockDim.x + threadIdx.x;
  if (i < NX) {
    const float* g = gbuf + i * SSTEPS;
    float acc = 0.0f;
    for (int s = 0; s < SSTEPS; ++s) acc = fmaf(g[s], ccw[s], acc);
    out[i] = acc * (xbuf[i] * 0.5f) + offset[0];
  }
}

extern "C" void kernel_launch(void* const* d_in, const int* in_sizes, int n_in,
                              void* d_out, int out_size, void* d_ws, size_t ws_size,
                              hipStream_t stream) {
  (void)in_sizes; (void)n_in; (void)out_size; (void)ws_size;
  const float* logits = (const float*)d_in[0];
  const float* W1     = (const float*)d_in[1];
  const float* b1     = (const float*)d_in[2];
  const float* W2     = (const float*)d_in[3];
  const float* b2     = (const float*)d_in[4];
  const float* W3     = (const float*)d_in[5];
  const float* b3     = (const float*)d_in[6];
  const float* offset = (const float*)d_in[7];
  float* out = (float*)d_out;

  /* scratch: x[10240] | ccw[40] | steps[40] | g[337920] | w2h f16[65536] */
  float* ws_x     = (float*)d_ws;
  float* ws_ccw   = ws_x + NX;
  float* ws_steps = ws_ccw + 40;
  float* ws_g     = ws_steps + 40;
  _Float16* ws_w2h = (_Float16*)(ws_g + MTOTAL);   /* byte off 1392960, 16B aligned */

  hipFuncSetAttribute(reinterpret_cast<const void*>(umnn_main_kernel),
                      hipFuncAttributeMaxDynamicSharedMemorySize, SMEM_BYTES);

  prep_kernel<<<PREP_THREADS / 256, 256, 0, stream>>>(logits, W2, ws_x, ws_ccw,
                                                      ws_steps, ws_w2h);
  umnn_main_kernel<<<MAIN_GRID, BLOCK_THREADS, SMEM_BYTES, stream>>>(
      W1, b1, ws_w2h, b2, W3, b3, ws_x, ws_steps, ws_g);
  finish_kernel<<<(NX + 255) / 256, 256, 0, stream>>>(ws_x, ws_g, ws_ccw, offset, out);
}
